// HybridMambaCNN_11854109737407
// MI455X (gfx1250) — matrix operand order repaired and checked by reading, not compiled, not run on hardware
//
#include <hip/hip_runtime.h>
#include <hip/hip_bf16.h>

typedef __attribute__((ext_vector_type(16))) __bf16 v16bf;
typedef __attribute__((ext_vector_type(8)))  float  v8f;

// ---------------------------------------------------------------------------
// WMMA helpers (CDNA5 gfx1250, wave32). Layouts per CDNA5 ISA 7.12.2:
//  A (16x32 bf16): lane m = lane&15, group g = lane>>4;
//                  element e -> k = e + 8*g + (e>=8 ? 8 : 0)
//  B (32x16 bf16): lane n = lane&15, group g = lane>>4; element e -> k = e + 8*g + (e>=8 ? 8 : 0)
//  C/D (16x16 f32): vgpr r, lane -> row = r + 8*g, col = lane&15
// ---------------------------------------------------------------------------

// constant per-element (post-unroll) part of the A-fragment k index; the
// runtime 8*gsel part is folded into the per-lane base pointer.
__device__ __forceinline__ constexpr int a_eoff(int e) { return e + ((e >= 8) ? 8 : 0); }

// Weights pre-swizzled into fragment-register layout:
//   Bp[ ((nt*ktiles + kt)*32 + lane) * 16 + e ] == W[k = kt*32+8*(lane>>4)+a_eoff(e)][n = nt*16+(lane&15)]
// -> one aligned 32-byte vector load per lane (2x global_load_b128).
__device__ __forceinline__ v16bf load_b_fragp(const __bf16* __restrict__ Bp,
                                              int ktiles, int kt, int nt, int lane) {
    return *(const v16bf*)(Bp + (((size_t)(nt * ktiles + kt) * 32 + lane) << 4));
}

// A fragment from a strided f32 matrix: base points at (row = k0 + 8*gsel, this
// lane's m-column); element e adds a compile-time row offset a_eoff(e)*stride.
template <int STRIDE>
__device__ __forceinline__ v16bf load_a_strided(const float* __restrict__ base) {
    v16bf a;
#pragma unroll
    for (int e = 0; e < 16; ++e) a[e] = (__bf16)base[a_eoff(e) * STRIDE];
    return a;
}

__device__ __forceinline__ v8f wmma_bf16(v16bf a, v16bf b, v8f c) {
    return __builtin_amdgcn_wmma_f32_16x16x32_bf16(
        /*neg_a=*/false, a, /*neg_b=*/false, b,
        /*c_mod=*/(short)0, c, /*reuse_a=*/false, /*reuse_b=*/false);
}

__device__ __forceinline__ float silu_f(float x) { return x / (1.0f + __expf(-x)); }

__device__ __forceinline__ void frag_coords(int i, int ktiles, int* k, int* n) {
    const int e = i & 15, lane = (i >> 4) & 31, t = i >> 9;
    const int kt = t % ktiles, nt = t / ktiles;
    *k = kt * 32 + 8 * (lane >> 4) + e + ((e >= 8) ? 8 : 0);
    *n = nt * 16 + (lane & 15);
}

// ---------------------------------------------------------------------------
// Problem constants
// ---------------------------------------------------------------------------
#define BATCH   128
#define C_IN    200
#define PDIM    32
#define LTOK    1024          // 32*32
#define D_MODEL 32
#define D_INNER 64
#define NSTATE  16
#define FDIM    128
#define M_TOK   (BATCH * LTOK)   // 131072

// ---------------------------------------------------------------------------
// Weight packing into fragment layout
// ---------------------------------------------------------------------------
__global__ void pack_w_frag_kernel(const float* __restrict__ W, __bf16* __restrict__ Bp,
                                   int K, int Kpad, int N) {
    const int ktiles = Kpad / 32;
    const int total = (N / 16) * ktiles * 512;
    int i = blockIdx.x * blockDim.x + threadIdx.x;
    if (i >= total) return;
    int k, n; frag_coords(i, ktiles, &k, &n);
    Bp[i] = (k < K) ? (__bf16)W[n * K + k] : (__bf16)0.0f;
}

// Fused x_proj / dt projection weights -> frag layout, logical 64x96:
//   cols [0,64):  Wd[k][d]  = sum_r x_proj_w[r][k] * dt_w[d][r]   (delta path)
//   cols [64,80): x_proj_w[2+n][k]                                 (B path)
//   cols [80,96): x_proj_w[18+n][k]                                (C path)
__global__ void pack_xproj_frag_kernel(const float* __restrict__ x_proj_w,
                                       const float* __restrict__ dt_w,
                                       __bf16* __restrict__ Bp) {
    const int ktiles = 2;                       // K = 64
    int i = blockIdx.x * blockDim.x + threadIdx.x;
    if (i >= 6 * ktiles * 512) return;          // N = 96
    int k, n; frag_coords(i, ktiles, &k, &n);
    float v;
    if (n < 64)       v = x_proj_w[0 * 64 + k] * dt_w[n * 2 + 0] +
                          x_proj_w[1 * 64 + k] * dt_w[n * 2 + 1];
    else if (n < 80)  v = x_proj_w[(2  + (n - 64)) * 64 + k];
    else              v = x_proj_w[(18 + (n - 80)) * 64 + k];
    Bp[i] = (__bf16)v;
}

// conv weights (Co,Ci,3,3), Ci == 32 -> frag layout with K reordered so each
// 32-wide K tile is one (kh,kw) tap:  k = (kh*3+kw)*32 + ci
__global__ void pack_conv_frag_kernel(const float* __restrict__ W, __bf16* __restrict__ Bp,
                                      int Cout) {
    const int ktiles = 9;                       // K = 9*32 = 288
    const int total = (Cout / 16) * ktiles * 512;
    int i = blockIdx.x * blockDim.x + threadIdx.x;
    if (i >= total) return;
    int k, n; frag_coords(i, ktiles, &k, &n);
    const int r = k >> 5, ci = k & 31;
    Bp[i] = (__bf16)W[(n * 32 + ci) * 9 + r];
}

// W2[k][f] = sum_j out_proj_w[j][k] * fc_w[f][j]   (64 x 128, f32 - tiny)
__global__ void w2_kernel(const float* __restrict__ out_proj_w,
                          const float* __restrict__ fc_w,
                          float* __restrict__ W2) {
    int i = blockIdx.x * blockDim.x + threadIdx.x;
    if (i >= 64 * 128) return;
    int k = i / 128, f = i % 128;
    float s = 0.0f;
#pragma unroll
    for (int j = 0; j < 32; ++j) s += out_proj_w[j * 64 + k] * fc_w[f * 32 + j];
    W2[i] = s;
}

// ---------------------------------------------------------------------------
// Stage 1: p = relu(bn0(x @ proj_w^T + proj_b)),  GEMM M=131072 K=200(->224) N=32
// ---------------------------------------------------------------------------
__global__ void proj_bn_kernel(const float* __restrict__ x, const __bf16* __restrict__ Wp,
                               const float* __restrict__ pb,
                               const float* __restrict__ g, const float* __restrict__ bta,
                               const float* __restrict__ mu, const float* __restrict__ var,
                               float* __restrict__ p) {
    const int lane = threadIdx.x & 31, wave = threadIdx.x >> 5;
    const int tile = blockIdx.x * 8 + wave;
    const int row0 = tile * 16;
    const int gsel = lane >> 4;
    const int m = row0 + (lane & 15);
    const int b = m >> 10, hw = m & 1023;
    // per-lane base: row k = 8*gsel of this m's column
    const float* xb = x + ((size_t)(b * C_IN + 8 * gsel)) * LTOK + hw;

    v8f acc0 = {0,0,0,0,0,0,0,0}, acc1 = {0,0,0,0,0,0,0,0};
#pragma unroll
    for (int kt = 0; kt < 7; ++kt) {
        const int k0 = kt * 32;
        v16bf a;
        if (kt < 6) {
            a = load_a_strided<LTOK>(xb + (size_t)k0 * LTOK);
        } else {      // tail tile: guard k = k0 + a_eoff(e) + 8*gsel < 200
#pragma unroll
            for (int e = 0; e < 16; ++e) {
                const int koff = k0 + a_eoff(e);
                a[e] = (koff + 8 * gsel < C_IN) ? (__bf16)xb[(size_t)koff * LTOK]
                                                : (__bf16)0.0f;
            }
        }
        acc0 = wmma_bf16(a, load_b_fragp(Wp, 7, kt, 0, lane), acc0);
        acc1 = wmma_bf16(a, load_b_fragp(Wp, 7, kt, 1, lane), acc1);
    }
#pragma unroll
    for (int r = 0; r < 8; ++r) {
        const int mr = row0 + r + 8 * gsel;
        const int b2 = mr >> 10, hw2 = mr & 1023;
#pragma unroll
        for (int t = 0; t < 2; ++t) {
            const int n = t * 16 + (lane & 15);
            float v = (t ? acc1[r] : acc0[r]) + pb[n];
            v = (v - mu[n]) * (g[n] * rsqrtf(var[n] + 1e-5f)) + bta[n];
            p[(b2 * D_MODEL + n) * LTOK + hw2] = fmaxf(v, 0.0f);
        }
    }
}

// ---------------------------------------------------------------------------
// Stage 2: xz = tok @ in_proj_w^T.  K=32 (single WMMA step), N=128.
// ---------------------------------------------------------------------------
__global__ void inproj_kernel(const float* __restrict__ p, const __bf16* __restrict__ Wip,
                              float* __restrict__ xm, float* __restrict__ sz) {
    const int lane = threadIdx.x & 31, wave = threadIdx.x >> 5;
    const int tile = blockIdx.x * 8 + wave;
    const int row0 = tile * 16;
    const int gsel = lane >> 4;
    const int m = row0 + (lane & 15);
    const int b = m >> 10, l = m & 1023;

    const v16bf a = load_a_strided<LTOK>(p + (size_t)(b * D_MODEL + 8 * gsel) * LTOK + l);

    v8f acc[8];
#pragma unroll
    for (int t = 0; t < 8; ++t) {
        acc[t] = (v8f){0,0,0,0,0,0,0,0};
        acc[t] = wmma_bf16(a, load_b_fragp(Wip, 1, 0, t, lane), acc[t]);
    }
#pragma unroll
    for (int r = 0; r < 8; ++r) {
        const int mr = row0 + r + 8 * gsel;
        const int b2 = mr >> 10, l2 = mr & 1023;
#pragma unroll
        for (int t = 0; t < 8; ++t) {
            const int n = t * 16 + (lane & 15);
            const float v = acc[t][r];
            if (n < D_INNER) xm[(b2 * D_INNER + n) * LTOK + l2] = v;
            else             sz[mr * D_INNER + (n - D_INNER)] = silu_f(v);
        }
    }
}

// ---------------------------------------------------------------------------
// Stage 3: depthwise causal conv1d(k=4) + SiLU.  xm (B,64,L) -> u (B,L,64)
// ---------------------------------------------------------------------------
__global__ void dwconv_silu_kernel(const float* __restrict__ xm,
                                   const float* __restrict__ cw,
                                   const float* __restrict__ cb,
                                   float* __restrict__ u) {
    int i = blockIdx.x * blockDim.x + threadIdx.x;
    if (i >= BATCH * D_INNER * LTOK) return;
    const int l  = i & 1023;
    const int t  = i >> 10;
    const int ch = t & 63;
    const int b  = t >> 6;
    const float* src = xm + (b * D_INNER + ch) * LTOK;
    float s = cb[ch];
#pragma unroll
    for (int j = 0; j < 4; ++j) {
        const int ll = l - 3 + j;
        if (ll >= 0) s += cw[ch * 4 + j] * src[ll];
    }
    u[(b * LTOK + l) * D_INNER + ch] = silu_f(s);
}

// ---------------------------------------------------------------------------
// Stage 4: fused [delta | B | C] = u @ Wx (64x96) ; softplus(+dt_b) on delta cols.
// A is row-major contiguous -> vectorized float4 loads (2x 32B runs per frag).
// ---------------------------------------------------------------------------
__global__ void xdbl_kernel(const float* __restrict__ u, const __bf16* __restrict__ Wx,
                            const float* __restrict__ dt_b,
                            float* __restrict__ delta, float* __restrict__ Bm,
                            float* __restrict__ Cm) {
    const int lane = threadIdx.x & 31, wave = threadIdx.x >> 5;
    const int tile = blockIdx.x * 8 + wave;
    const int row0 = tile * 16;
    const int gsel = lane >> 4;
    const int m = row0 + (lane & 15);

    v8f acc[6];
#pragma unroll
    for (int t = 0; t < 6; ++t) acc[t] = (v8f){0,0,0,0,0,0,0,0};

#pragma unroll
    for (int kt = 0; kt < 2; ++kt) {
        const int k0 = kt * 32;
        // e<8: k = k0+8g+e (contiguous 8 floats); e>=8: k = k0+16+8g+(e-8)
        const float4* s0 = (const float4*)(u + m * D_INNER + k0 + 8 * gsel);
        const float4* s1 = (const float4*)(u + m * D_INNER + k0 + 16 + 8 * gsel);
        const float4 f0 = s0[0], f1 = s0[1], f2 = s1[0], f3 = s1[1];
        v16bf a;
        a[0]=(__bf16)f0.x;  a[1]=(__bf16)f0.y;  a[2]=(__bf16)f0.z;  a[3]=(__bf16)f0.w;
        a[4]=(__bf16)f1.x;  a[5]=(__bf16)f1.y;  a[6]=(__bf16)f1.z;  a[7]=(__bf16)f1.w;
        a[8]=(__bf16)f2.x;  a[9]=(__bf16)f2.y;  a[10]=(__bf16)f2.z; a[11]=(__bf16)f2.w;
        a[12]=(__bf16)f3.x; a[13]=(__bf16)f3.y; a[14]=(__bf16)f3.z; a[15]=(__bf16)f3.w;
#pragma unroll
        for (int t = 0; t < 6; ++t)
            acc[t] = wmma_bf16(a, load_b_fragp(Wx, 2, kt, t, lane), acc[t]);
    }
#pragma unroll
    for (int r = 0; r < 8; ++r) {
        const int mr = row0 + r + 8 * gsel;
#pragma unroll
        for (int t = 0; t < 6; ++t) {
            const int n = t * 16 + (lane & 15);
            float v = acc[t][r];
            if (n < 64) {
                v += dt_b[n];
                v = (v > 20.0f) ? v : log1pf(__expf(v));   // softplus
                delta[mr * 64 + n] = v;
            } else if (n < 80) {
                Bm[mr * NSTATE + (n - 64)] = v;
            } else {
                Cm[mr * NSTATE + (n - 80)] = v;
            }
        }
    }
}

// ---------------------------------------------------------------------------
// Stage 5: selective scan. One thread per (b,d) chain (8192 chains), 16 states
// in registers. Fuses +u*D and *silu(z). Writes y in place over u.
// ---------------------------------------------------------------------------
__global__ void scan_kernel(const float* __restrict__ delta, const float* __restrict__ Bm,
                            const float* __restrict__ Cm, float* __restrict__ u_y,
                            const float* __restrict__ sz, const float* __restrict__ A_log,
                            const float* __restrict__ Dv) {
    const int i = blockIdx.x * blockDim.x + threadIdx.x;
    if (i >= BATCH * D_INNER) return;
    const int d = i & 63, b = i >> 6;

    float A[NSTATE], h[NSTATE];
#pragma unroll
    for (int n = 0; n < NSTATE; ++n) { A[n] = -__expf(A_log[d * NSTATE + n]); h[n] = 0.0f; }
    const float Dd = Dv[d];

    for (int l = 0; l < LTOK; ++l) {
        const int rowv = (b * LTOK + l) * D_INNER + d;
        const int rown = (b * LTOK + l) * NSTATE;
        const float dl = delta[rowv];
        const float ut = u_y[rowv];
        if (l + 1 < LTOK) {
            __builtin_prefetch(&u_y[rowv + D_INNER], 0, 3);
            __builtin_prefetch(&delta[rowv + D_INNER], 0, 3);
        }
        const float du = dl * ut;
        float y = 0.0f;
#pragma unroll
        for (int n = 0; n < NSTATE; ++n) {
            const float dA = __expf(dl * A[n]);
            h[n] = dA * h[n] + du * Bm[rown + n];
            y += h[n] * Cm[rown + n];
        }
        y += ut * Dd;
        u_y[rowv] = y * sz[rowv];
    }
}

// ---------------------------------------------------------------------------
// Stage 6: ybar = mean_l y  (128 x 64), then out = ybar @ W2 + fc_b
// ---------------------------------------------------------------------------
__global__ void ybar_kernel(const float* __restrict__ y, float* __restrict__ ybar) {
    const int b = blockIdx.x, k = threadIdx.x;   // 128 blocks x 64 threads
    float s = 0.0f;
    const float* row = y + b * LTOK * D_INNER + k;
    for (int l = 0; l < LTOK; ++l) s += row[l * D_INNER];
    ybar[b * D_INNER + k] = s * (1.0f / (float)LTOK);
}

__global__ void mamba_out_kernel(const float* __restrict__ ybar, const float* __restrict__ W2,
                                 const float* __restrict__ fc_b, float* __restrict__ out) {
    const int i = blockIdx.x * blockDim.x + threadIdx.x;
    if (i >= BATCH * FDIM) return;
    const int b = i >> 7, f = i & 127;
    float s = fc_b[f];
#pragma unroll
    for (int k = 0; k < D_INNER; ++k) s += ybar[b * D_INNER + k] * W2[k * FDIM + f];
    out[i] = s;
}

// ---------------------------------------------------------------------------
// CNN branch: implicit-GEMM 3x3 VALID convs with WMMA.
// K reordered as k = (kh*3+kw)*32 + ci so each K-tile is one conv tap:
// A-fragment = base pointer + compile-time channel-stride offsets.
// ---------------------------------------------------------------------------
__global__ void conv1_kernel(const float* __restrict__ p, const __bf16* __restrict__ Wc,
                             const float* __restrict__ cb,
                             const float* __restrict__ g, const float* __restrict__ bta,
                             const float* __restrict__ mu, const float* __restrict__ var,
                             float* __restrict__ outb) {
    const int lane = threadIdx.x & 31, wave = threadIdx.x >> 5;
    const int tile = blockIdx.x * 8 + wave;
    const int row0 = tile * 16;
    const int gsel = lane >> 4;
    const int m = row0 + (lane & 15);
    const int b = m / 900, s0 = m % 900;
    const int oh = s0 / 30, ow = s0 % 30;
    const float* src = p + (size_t)(b * D_MODEL + 8 * gsel) * LTOK + oh * PDIM + ow;

    v8f acc0 = {0,0,0,0,0,0,0,0}, acc1 = {0,0,0,0,0,0,0,0};
#pragma unroll
    for (int kt = 0; kt < 9; ++kt) {
        const int kh = kt / 3, kw = kt % 3;
        const v16bf a = load_a_strided<LTOK>(src + kh * PDIM + kw);
        acc0 = wmma_bf16(a, load_b_fragp(Wc, 9, kt, 0, lane), acc0);
        acc1 = wmma_bf16(a, load_b_fragp(Wc, 9, kt, 1, lane), acc1);
    }
#pragma unroll
    for (int r = 0; r < 8; ++r) {
        const int mr = row0 + r + 8 * gsel;
        const int b2 = mr / 900, s2 = mr % 900;
#pragma unroll
        for (int t = 0; t < 2; ++t) {
            const int n = t * 16 + (lane & 15);
            float v = (t ? acc1[r] : acc0[r]) + cb[n];
            v = (v - mu[n]) * (g[n] * rsqrtf(var[n] + 1e-5f)) + bta[n];
            outb[(b2 * 32 + n) * 900 + s2] = fmaxf(v, 0.0f);
        }
    }
}

// conv2: c1out (B,32,30,30) -> bn2+relu -> spatial-mean accumulated into out (+=)
__global__ void conv2_kernel(const float* __restrict__ p2, const __bf16* __restrict__ Wc,
                             const float* __restrict__ cb,
                             const float* __restrict__ g, const float* __restrict__ bta,
                             const float* __restrict__ mu, const float* __restrict__ var,
                             float* __restrict__ out) {
    const int lane = threadIdx.x & 31, wave = threadIdx.x >> 5;
    const int tile = blockIdx.x * 8 + wave;
    const int row0 = tile * 16;
    const int gsel = lane >> 4;
    const int m = row0 + (lane & 15);
    const int b = m / 784, s0 = m % 784;
    const int oh = s0 / 28, ow = s0 % 28;
    const float* src = p2 + (size_t)(b * 32 + 8 * gsel) * 900 + oh * 30 + ow;

    v8f acc[8];
#pragma unroll
    for (int t = 0; t < 8; ++t) acc[t] = (v8f){0,0,0,0,0,0,0,0};

#pragma unroll 3
    for (int kt = 0; kt < 9; ++kt) {
        const int kh = kt / 3, kw = kt % 3;
        const v16bf a = load_a_strided<900>(src + kh * 30 + kw);
#pragma unroll
        for (int t = 0; t < 8; ++t)
            acc[t] = wmma_bf16(a, load_b_fragp(Wc, 9, kt, t, lane), acc[t]);
    }
#pragma unroll
    for (int r = 0; r < 8; ++r) {
        const int mr = row0 + r + 8 * gsel;
        const int b2 = mr / 784;
#pragma unroll
        for (int t = 0; t < 8; ++t) {
            const int n = t * 16 + (lane & 15);
            float v = acc[t][r] + cb[n];
            v = (v - mu[n]) * (g[n] * rsqrtf(var[n] + 1e-5f)) + bta[n];
            v = fmaxf(v, 0.0f);
            atomicAdd(&out[b2 * FDIM + n], v * (1.0f / 784.0f));
        }
    }
}

// ---------------------------------------------------------------------------
// Host launch
// ---------------------------------------------------------------------------
extern "C" void kernel_launch(void* const* d_in, const int* in_sizes, int n_in,
                              void* d_out, int out_size, void* d_ws, size_t ws_size,
                              hipStream_t stream) {
    const float* x          = (const float*)d_in[0];
    const float* proj_w     = (const float*)d_in[1];
    const float* proj_b     = (const float*)d_in[2];
    const float* bn0_g      = (const float*)d_in[3];
    const float* bn0_b      = (const float*)d_in[4];
    const float* bn0_m      = (const float*)d_in[5];
    const float* bn0_v      = (const float*)d_in[6];
    const float* in_proj_w  = (const float*)d_in[7];
    const float* conv_w     = (const float*)d_in[8];
    const float* conv_b     = (const float*)d_in[9];
    const float* x_proj_w   = (const float*)d_in[10];
    const float* dt_w       = (const float*)d_in[11];
    const float* dt_b       = (const float*)d_in[12];
    const float* A_log      = (const float*)d_in[13];
    const float* Dv         = (const float*)d_in[14];
    const float* out_proj_w = (const float*)d_in[15];
    const float* fc_w       = (const float*)d_in[16];
    const float* fc_b       = (const float*)d_in[17];
    const float* c1_w       = (const float*)d_in[18];
    const float* c1_b       = (const float*)d_in[19];
    const float* bn1_g      = (const float*)d_in[20];
    const float* bn1_b      = (const float*)d_in[21];
    const float* bn1_m      = (const float*)d_in[22];
    const float* bn1_v      = (const float*)d_in[23];
    const float* c2_w       = (const float*)d_in[24];
    const float* c2_b       = (const float*)d_in[25];
    const float* bn2_g      = (const float*)d_in[26];
    const float* bn2_b      = (const float*)d_in[27];
    const float* bn2_m      = (const float*)d_in[28];
    const float* bn2_v      = (const float*)d_in[29];
    float* out = (float*)d_out;

    // ---- workspace carve-up (256B aligned) ----
    char* ws = (char*)d_ws;
    size_t off = 0;
    auto alloc = [&](size_t bytes) { void* p = ws + off; off = (off + bytes + 255) & ~(size_t)255; return p; };

    __bf16* Wp  = (__bf16*)alloc(224 * 32  * sizeof(__bf16)); // proj (frag layout)
    __bf16* Wip = (__bf16*)alloc(32  * 128 * sizeof(__bf16)); // in_proj
    __bf16* Wx  = (__bf16*)alloc(64  * 96  * sizeof(__bf16)); // fused xproj/dt
    __bf16* Wc1 = (__bf16*)alloc(288 * 32  * sizeof(__bf16));
    __bf16* Wc2 = (__bf16*)alloc(288 * 128 * sizeof(__bf16));
    float*  W2  = (float*) alloc(64  * 128 * sizeof(float));
    float*  p   = (float*) alloc((size_t)BATCH * D_MODEL * LTOK * sizeof(float));   // 16 MB
    float*  xm  = (float*) alloc((size_t)BATCH * D_INNER * LTOK * sizeof(float));   // 32 MB (c1out aliases)
    float*  sz  = (float*) alloc((size_t)M_TOK * D_INNER * sizeof(float));          // 32 MB
    float*  u   = (float*) alloc((size_t)M_TOK * D_INNER * sizeof(float));          // 32 MB (y in place)
    float*  dl  = (float*) alloc((size_t)M_TOK * D_INNER * sizeof(float));          // 32 MB
    float*  Bm  = (float*) alloc((size_t)M_TOK * NSTATE  * sizeof(float));          // 8 MB
    float*  Cm  = (float*) alloc((size_t)M_TOK * NSTATE  * sizeof(float));          // 8 MB
    float*  yb  = (float*) alloc((size_t)BATCH * D_INNER * sizeof(float));
    float*  c1o = xm;  // alias: xm is dead after dwconv; 128*32*900*4 = 14.7MB < 32MB

    // ---- weight preparation (tiny) ----
    pack_w_frag_kernel<<<(224 * 32 + 255) / 256, 256, 0, stream>>>(proj_w, Wp, 200, 224, 32);
    pack_w_frag_kernel<<<(32 * 128 + 255) / 256, 256, 0, stream>>>(in_proj_w, Wip, 32, 32, 128);
    pack_xproj_frag_kernel<<<(64 * 96 + 255) / 256, 256, 0, stream>>>(x_proj_w, dt_w, Wx);
    pack_conv_frag_kernel<<<(288 * 32 + 255) / 256, 256, 0, stream>>>(c1_w, Wc1, 32);
    pack_conv_frag_kernel<<<(288 * 128 + 255) / 256, 256, 0, stream>>>(c2_w, Wc2, 128);
    w2_kernel<<<(64 * 128 + 255) / 256, 256, 0, stream>>>(out_proj_w, fc_w, W2);

    // ---- mamba branch ----
    proj_bn_kernel<<<M_TOK / 128, 256, 0, stream>>>(x, Wp, proj_b, bn0_g, bn0_b, bn0_m, bn0_v, p);
    inproj_kernel<<<M_TOK / 128, 256, 0, stream>>>(p, Wip, xm, sz);
    dwconv_silu_kernel<<<(BATCH * D_INNER * LTOK + 255) / 256, 256, 0, stream>>>(xm, conv_w, conv_b, u);
    xdbl_kernel<<<M_TOK / 128, 256, 0, stream>>>(u, Wx, dt_b, dl, Bm, Cm);
    scan_kernel<<<(BATCH * D_INNER + 255) / 256, 256, 0, stream>>>(dl, Bm, Cm, u, sz, A_log, Dv);
    ybar_kernel<<<BATCH, D_INNER, 0, stream>>>(u, yb);
    mamba_out_kernel<<<(BATCH * FDIM + 255) / 256, 256, 0, stream>>>(yb, W2, fc_b, out);

    // ---- CNN branch (conv2 accumulates spatial mean into out) ----
    conv1_kernel<<<(BATCH * 900) / 128, 256, 0, stream>>>(p, Wc1, c1_b, bn1_g, bn1_b, bn1_m, bn1_v, c1o);
    conv2_kernel<<<(BATCH * 784) / 128, 256, 0, stream>>>(c1o, Wc2, c2_b, bn2_g, bn2_b, bn2_m, bn2_v, out);

    (void)in_sizes; (void)n_in; (void)out_size; (void)ws_size;
}
